// QGRU_22771916603952
// MI455X (gfx1250) — compile-verified, matrix operand order checked by reading, not run on hardware
//
#include <hip/hip_runtime.h>

// ---------------- problem constants ----------------
#define T_STEPS 500
#define BATCH   16
#define D_IN    512
#define HID     1024
#define KT0     (D_IN / 32)   // 16 K-tiles for layer-0 input projection
#define KT1     (HID / 32)    // 32 K-tiles for H-sized GEMMs
#define NTILES  (HID / 16)    // 64 N-tiles of width 16
#define REC_NWG 32            // persistent WGs in recurrence (2 waves, 32 cols each)
#define REC_LDS_BYTES (2 * 3 * KT1 * 512 * 2)   // 2 waves x 3 gates x 32 frags x 1KB = 192KB

typedef __attribute__((ext_vector_type(16))) __bf16 bf16x16;
typedef __attribute__((ext_vector_type(8)))  float  floatx8;
typedef __attribute__((ext_vector_type(4)))  unsigned int uint32x4;
typedef __attribute__((ext_vector_type(8)))  int int32x8;
typedef __attribute__((ext_vector_type(4)))  int int32x4;
typedef __bf16 bf16_t;

// float -> bf16 round-to-nearest-even
__device__ __forceinline__ bf16_t f2bf(float f) {
    union { float f; unsigned u; } v; v.f = f;
    unsigned r = v.u + 0x7FFFu + ((v.u >> 16) & 1u);
    unsigned short h = (unsigned short)(r >> 16);
    bf16_t out; __builtin_memcpy(&out, &h, 2);
    return out;
}

__device__ __forceinline__ floatx8 wmma_bf16(bf16x16 a, bf16x16 b, floatx8 c) {
    return __builtin_amdgcn_wmma_f32_16x16x32_bf16(false, a, false, b,
                                                   (short)0, c, false, false);
}

__device__ __forceinline__ float sigmoidf_(float x) {
    return 1.0f / (1.0f + __expf(-x));
}

// ---------------------------------------------------------------------------
// Fragment layouts (CDNA5 ISA 7.12.2, 16-bit, wave32):
//   A 16x32: lane = m + 16*khalf ; slot s: g=s>>3, w=s&7, k_local=g*16+khalf*8+w
//   B 32x16: mirrored (lane = n + 16*khalf, same slot->k mapping)
// Packed: element = fragIndex*512 + lane*16 + slot.
//   proj B frags:      fragIndex = nt*KT + kt
//   recurrence U frags: fragIndex = (nt*3 + gate)*KT + kt   (gate-major per N-tile
//   so one contiguous 96KB TDM tile stages a wave's full 3-gate slice into LDS)
// ---------------------------------------------------------------------------

__device__ __forceinline__ float qw_elem(const float* __restrict__ w,
                                         int k, int n, int inq, int outq) {
    const int comp_t[4][4] = {{0,1,2,3},{1,0,3,2},{2,3,0,1},{3,2,1,0}};
    const int sign_t[4][4] = {{1,-1,-1,-1},{1,1,-1,1},{1,1,1,-1},{1,-1,1,1}};
    int rb = k / inq, a = k % inq;
    int cb = n / outq, b = n % outq;
    int c = comp_t[cb][rb];
    float v = w[(size_t)c * inq * outq + (size_t)a * outq + b];
    return (sign_t[cb][rb] > 0) ? v : -v;
}

// Assemble quaternion (Hamilton block) matrix + pack to B-fragment layout.
__global__ void pack_qw_kernel(const float* __restrict__ w, bf16_t* __restrict__ dst,
                               int inq, int outq, int KT, int ntStride, int gateOff) {
    int id = blockIdx.x * blockDim.x + threadIdx.x;
    int total = inq * 4 * outq * 4;
    if (id >= total) return;
    int slot = id & 15;
    int lane = (id >> 4) & 31;
    int frag = id >> 9;
    int kt = frag % KT;
    int nt = frag / KT;
    int khalf = lane >> 4, nl = lane & 15;
    int g = slot >> 3, wbit = slot & 7;
    int k = kt * 32 + g * 16 + khalf * 8 + wbit;
    int n = nt * 16 + nl;
    size_t didx = (((size_t)nt * ntStride + gateOff + kt) << 9) + lane * 16 + slot;
    dst[didx] = f2bf(qw_elem(w, k, n, inq, outq));
}

// Pack x[T][B][D] (f32) into A-fragment layout (bf16); mt = t, m = b.
__global__ void pack_x_kernel(const float* __restrict__ x, bf16_t* __restrict__ dst) {
    int id = blockIdx.x * blockDim.x + threadIdx.x;
    const int total = T_STEPS * KT0 * 512;
    if (id >= total) return;
    int slot = id & 15;
    int lane = (id >> 4) & 31;
    int frag = id >> 9;
    int kt = frag % KT0;
    int mt = frag / KT0;
    int khalf = lane >> 4, m = lane & 15;
    int g = slot >> 3, wbit = slot & 7;
    int k = kt * 32 + g * 16 + khalf * 8 + wbit;
    dst[id] = f2bf(x[(size_t)mt * BATCH * D_IN + (size_t)m * D_IN + k]);
}

// Input-projection GEMM, double-buffered over kt so loads for kt+1 overlap the
// wmmas of kt (partial s_wait_loadcnt instead of wait-0 before every wmma).
// grid = (T_STEPS, 4, 3 gates), block = 128 (4 waves x 4 N-tiles, A reuse x4)
__global__ void __launch_bounds__(128)
proj_kernel(const bf16_t* __restrict__ Ap,
            const bf16_t* __restrict__ Bz, const bf16_t* __restrict__ Br,
            const bf16_t* __restrict__ Bh,
            const float* __restrict__ bz, const float* __restrict__ br,
            const float* __restrict__ bh,
            float* __restrict__ oz, float* __restrict__ orr,
            float* __restrict__ oh, int KT) {
    int wave = threadIdx.x >> 5;
    int lane = threadIdx.x & 31;
    int mt   = blockIdx.x;
    int gate = blockIdx.z;
    const bf16_t* Bp   = (gate == 0) ? Bz : (gate == 1) ? Br : Bh;
    const float*  bias = (gate == 0) ? bz : (gate == 1) ? br : bh;
    float*        out  = (gate == 0) ? oz : (gate == 1) ? orr : oh;

    int nt0 = (blockIdx.y * 4 + wave) * 4;
    floatx8 acc[4] = {};
    const bf16_t* arow = Ap + ((size_t)mt * KT) * 512 + lane * 16;
    const bf16_t* bcol = Bp + lane * 16;

    bf16x16 a_cur = *(const bf16x16*)(arow);
    bf16x16 b_cur[4];
#pragma unroll
    for (int q = 0; q < 4; ++q)
        b_cur[q] = *(const bf16x16*)(bcol + (size_t)(nt0 + q) * KT * 512);

#pragma unroll 2
    for (int kt = 0; kt < KT - 1; ++kt) {
        bf16x16 a_nxt = *(const bf16x16*)(arow + (size_t)(kt + 1) * 512);
        bf16x16 b_nxt[4];
#pragma unroll
        for (int q = 0; q < 4; ++q)
            b_nxt[q] = *(const bf16x16*)(bcol + ((size_t)(nt0 + q) * KT + kt + 1) * 512);
#pragma unroll
        for (int q = 0; q < 4; ++q)
            acc[q] = wmma_bf16(a_cur, b_cur[q], acc[q]);
        a_cur = a_nxt;
#pragma unroll
        for (int q = 0; q < 4; ++q) b_cur[q] = b_nxt[q];
    }
#pragma unroll
    for (int q = 0; q < 4; ++q)
        acc[q] = wmma_bf16(a_cur, b_cur[q], acc[q]);

    int nl = lane & 15, mb = (lane >> 4) << 3;
#pragma unroll
    for (int q = 0; q < 4; ++q) {
#pragma unroll
        for (int d = 0; d < 8; ++d) {
            int n = (nt0 + q) * 16 + nl;
            out[(size_t)(mt * 16 + mb + d) * HID + n] = acc[q][d] + bias[n];
        }
    }
}

__global__ void init_rec_kernel(float* __restrict__ h, bf16_t* __restrict__ hp,
                                unsigned* __restrict__ bar) {
    int id = blockIdx.x * blockDim.x + threadIdx.x;
    int stride = gridDim.x * blockDim.x;
    for (int i = id; i < BATCH * HID; i += stride) h[i] = 0.0f;
    for (int i = id; i < KT1 * 512; i += stride) hp[i] = f2bf(0.0f);
    if (id < 4) bar[id] = 0u;
}

// Grid barrier: release fence before arrive, acquire fence (per-CU L0
// invalidate) after release so cross-WG hp/rhp fragments are seen. Weights
// live in LDS and are unaffected by the invalidate.
__device__ __forceinline__ void grid_barrier(unsigned* bar, unsigned target) {
    __threadfence();
    __syncthreads();
    if (threadIdx.x == 0) {
        if (atomicAdd(&bar[0], 1u) == (unsigned)(REC_NWG - 1)) {
            atomicExch(&bar[0], 0u);
            __hip_atomic_store(&bar[1], target, __ATOMIC_RELEASE,
                               __HIP_MEMORY_SCOPE_AGENT);
        } else {
            while (__hip_atomic_load(&bar[1], __ATOMIC_ACQUIRE,
                                     __HIP_MEMORY_SCOPE_AGENT) != target) {
                __builtin_amdgcn_s_sleep(1);
            }
        }
    }
    __syncthreads();
    __threadfence();   // acquire side: every wave invalidates its L0
}

// Persistent recurrence. 32 WGs x 2 waves; wave owns one 16-col N-tile.
// U-slices staged into LDS once via TDM (tensor_load_to_lds), then the 500-step
// loop reads weights from LDS (ds_load_b128) and only h/rh fragments from L2.
__global__ void __launch_bounds__(64)
rec_kernel(const float* __restrict__ pz, const float* __restrict__ pr,
           const float* __restrict__ ph,
           const bf16_t* __restrict__ Urec,   // [nt][gate][kt] fragment-packed
           float* __restrict__ h, bf16_t* __restrict__ hp,
           bf16_t* __restrict__ rhp,
           bf16_t* __restrict__ outA, float* __restrict__ outF,
           unsigned* __restrict__ bar) {
    extern __shared__ bf16_t ldsbuf[];
    const int KT = KT1;
    int wave = threadIdx.x >> 5;
    int lane = threadIdx.x & 31;
    int nt = blockIdx.x * 2 + wave;          // 0..63
    int nl = lane & 15, mb = (lane >> 4) << 3;
    int n = nt * 16 + nl;
    // A-layout scatter coords for column n (K index n in the next GEMM)
    int kt2 = n >> 5;
    int kl  = n & 31;
    int gA = kl >> 4, wA = kl & 15;
    int khalfA = wA >> 3;
    int slotA  = gA * 8 + (wA & 7);

    // ---- stage this wave's 96KB (3-gate) U-slice into LDS via TDM ----
    const unsigned sliceElems = 3u * KT * 512u;            // bf16 elements
    const unsigned sliceBytes = sliceElems * 2u;           // 98304
    unsigned lds_off = ((unsigned)(size_t)&ldsbuf[0]) + (unsigned)wave * sliceBytes;
    unsigned long long ga =
        (unsigned long long)(size_t)(Urec + (size_t)nt * sliceElems);
    const unsigned elems8 = sliceBytes / 8u;               // 12288 8-byte elems
    uint32x4 g0;
    g0[0] = 1u;                                            // count=1 (user D#)
    g0[1] = lds_off;                                       // lds_addr (bytes)
    g0[2] = (unsigned)(ga & 0xFFFFFFFFu);                  // global_addr[31:0]
    g0[3] = (unsigned)((ga >> 32) & 0x1FFFFFFu) | (2u << 30); // addr[56:32]|type=2
    int32x8 g1;
    g1[0] = (int)(3u << 16);                               // data_size=8B
    g1[1] = (int)((elems8 & 0xFFFFu) << 16);               // tensor_dim0[15:0]
    g1[2] = (int)((elems8 >> 16) | (1u << 16));            // dim0 hi | tensor_dim1=1
    g1[3] = (int)((elems8 & 0xFFFFu) << 16);               // tile_dim0 = whole row
    g1[4] = 1;                                             // tile_dim1 = 1
    g1[5] = (int)elems8;                                   // tensor_dim0_stride
    g1[6] = 0; g1[7] = 0;
    int32x4 gz4 = {0, 0, 0, 0};
    int32x8 gz8 = {0, 0, 0, 0, 0, 0, 0, 0};
    // 6-arg form (clang-23 / therock headers): groups 0..3 + extra group + cpol
    __builtin_amdgcn_tensor_load_to_lds(g0, g1, gz4, gz4, gz8, 0);
    __builtin_amdgcn_s_wait_tensorcnt(0);

    const bf16_t* lz = ldsbuf + (size_t)wave * sliceElems + lane * 16;
    const bf16_t* lr = lz + (size_t)KT * 512;
    const bf16_t* lh = lz + (size_t)2 * KT * 512;

    unsigned target = 0;
    for (int t = 0; t < T_STEPS; ++t) {
        // ---- phase A: z and r gates (h @ Uz, h @ Ur) ----
        floatx8 az = {}, ar = {};
        bf16x16 a_cur = *(const bf16x16*)(hp + lane * 16);
#pragma unroll
        for (int kt = 0; kt < KT; ++kt) {
            bf16x16 a_nxt;
            if (kt + 1 < KT)
                a_nxt = *(const bf16x16*)(hp + (size_t)(kt + 1) * 512 + lane * 16);
            bf16x16 bz = *(const bf16x16*)(lz + (size_t)kt * 512);
            bf16x16 br = *(const bf16x16*)(lr + (size_t)kt * 512);
            az = wmma_bf16(a_cur, bz, az);
            ar = wmma_bf16(a_cur, br, ar);
            a_cur = a_nxt;
        }
        const float* pzt = pz + (size_t)t * BATCH * HID;
        const float* prt = pr + (size_t)t * BATCH * HID;
        float zreg[8], hreg[8];
#pragma unroll
        for (int d = 0; d < 8; ++d) {
            int m = mb + d;
            float zv = sigmoidf_(pzt[(size_t)m * HID + n] + az[d]);
            float rv = sigmoidf_(prt[(size_t)m * HID + n] + ar[d]);
            float hv = h[(size_t)m * HID + n];
            zreg[d] = zv; hreg[d] = hv;
            int la = m + khalfA * 16;
            rhp[(size_t)kt2 * 512 + la * 16 + slotA] = f2bf(rv * hv);
        }
        grid_barrier(bar, ++target);

        // ---- phase B: candidate + state update ((r*h) @ Uh) ----
        floatx8 ah = {};
        bf16x16 c_cur = *(const bf16x16*)(rhp + lane * 16);
#pragma unroll
        for (int kt = 0; kt < KT; ++kt) {
            bf16x16 c_nxt;
            if (kt + 1 < KT)
                c_nxt = *(const bf16x16*)(rhp + (size_t)(kt + 1) * 512 + lane * 16);
            bf16x16 bh = *(const bf16x16*)(lh + (size_t)kt * 512);
            ah = wmma_bf16(c_cur, bh, ah);
            c_cur = c_nxt;
        }
        const float* pht = ph + (size_t)t * BATCH * HID;
#pragma unroll
        for (int d = 0; d < 8; ++d) {
            int m = mb + d;
            float hc = tanhf(pht[(size_t)m * HID + n] + ah[d]);
            float hn = zreg[d] * hreg[d] + (1.0f - zreg[d]) * hc;
            h[(size_t)m * HID + n] = hn;
            int la = m + khalfA * 16;
            bf16_t hb = f2bf(hn);
            hp[(size_t)kt2 * 512 + la * 16 + slotA] = hb;
            if (outA) outA[((size_t)t * KT1 + kt2) * 512 + la * 16 + slotA] = hb;
            if (outF) outF[((size_t)t * BATCH + m) * HID + n] = hn;
        }
        grid_barrier(bar, ++target);
    }
}

// ---------------------------------------------------------------------------
extern "C" void kernel_launch(void* const* d_in, const int* in_sizes, int n_in,
                              void* d_out, int out_size, void* d_ws, size_t ws_size,
                              hipStream_t stream) {
    const float* x = (const float*)d_in[0];
    const float* Wsrc0[3] = {(const float*)d_in[1], (const float*)d_in[3], (const float*)d_in[5]};
    const float* Bias0[3] = {(const float*)d_in[2], (const float*)d_in[4], (const float*)d_in[6]};
    const float* Usrc0[3] = {(const float*)d_in[7], (const float*)d_in[8], (const float*)d_in[9]};
    const float* Wsrc1[3] = {(const float*)d_in[10], (const float*)d_in[12], (const float*)d_in[14]};
    const float* Bias1[3] = {(const float*)d_in[11], (const float*)d_in[13], (const float*)d_in[15]};
    const float* Usrc1[3] = {(const float*)d_in[16], (const float*)d_in[17], (const float*)d_in[18]};

    char* ws = (char*)d_ws;
    size_t off = 0;
    auto take = [&](size_t bytes) -> char* {
        off = (off + 255) & ~(size_t)255;
        char* p = ws + off;
        off += bytes;
        return p;
    };

    bf16_t* x_p = (bf16_t*)take((size_t)T_STEPS * KT0 * 512 * 2);
    bf16_t *W0p[3], *W1p[3];
    for (int i = 0; i < 3; ++i) {
        W0p[i] = (bf16_t*)take((size_t)D_IN * HID * 2);
        W1p[i] = (bf16_t*)take((size_t)HID * HID * 2);
    }
    bf16_t* Urec0 = (bf16_t*)take((size_t)3 * HID * HID * 2);  // gate-major per nt
    bf16_t* Urec1 = (bf16_t*)take((size_t)3 * HID * HID * 2);
    float* pjz = (float*)take((size_t)T_STEPS * BATCH * HID * 4);
    float* pjr = (float*)take((size_t)T_STEPS * BATCH * HID * 4);
    float* pjh = (float*)take((size_t)T_STEPS * BATCH * HID * 4);
    bf16_t* out0p = (bf16_t*)take((size_t)T_STEPS * KT1 * 512 * 2);
    float*  hbuf  = (float*)take((size_t)BATCH * HID * 4);
    bf16_t* hp    = (bf16_t*)take((size_t)KT1 * 512 * 2);
    bf16_t* rhp   = (bf16_t*)take((size_t)KT1 * 512 * 2);
    unsigned* bar = (unsigned*)take(256);

    // 1) pack activations + quaternion weights to WMMA fragment layout (bf16)
    {
        int total = T_STEPS * KT0 * 512;
        pack_x_kernel<<<(total + 255) / 256, 256, 0, stream>>>(x, x_p);
    }
    for (int i = 0; i < 3; ++i) {
        pack_qw_kernel<<<(D_IN * HID + 255) / 256, 256, 0, stream>>>(
            Wsrc0[i], W0p[i], D_IN / 4, HID / 4, KT0, KT0, 0);
        pack_qw_kernel<<<(HID * HID + 255) / 256, 256, 0, stream>>>(
            Wsrc1[i], W1p[i], HID / 4, HID / 4, KT1, KT1, 0);
        pack_qw_kernel<<<(HID * HID + 255) / 256, 256, 0, stream>>>(
            Usrc0[i], Urec0, HID / 4, HID / 4, KT1, 3 * KT1, i * KT1);
        pack_qw_kernel<<<(HID * HID + 255) / 256, 256, 0, stream>>>(
            Usrc1[i], Urec1, HID / 4, HID / 4, KT1, 3 * KT1, i * KT1);
    }

    dim3 pgrid(T_STEPS, 4, 3);
    // 2) layer-0 input projections
    proj_kernel<<<pgrid, 128, 0, stream>>>(x_p, W0p[0], W0p[1], W0p[2],
                                           Bias0[0], Bias0[1], Bias0[2],
                                           pjz, pjr, pjh, KT0);
    // 3) layer-0 recurrence (persistent, TDM-staged weights in LDS)
    init_rec_kernel<<<64, 256, 0, stream>>>(hbuf, hp, bar);
    rec_kernel<<<REC_NWG, 64, REC_LDS_BYTES, stream>>>(
        pjz, pjr, pjh, Urec0, hbuf, hp, rhp, out0p, nullptr, bar);
    // 4) layer-1 input projections
    proj_kernel<<<pgrid, 128, 0, stream>>>(out0p, W1p[0], W1p[1], W1p[2],
                                           Bias1[0], Bias1[1], Bias1[2],
                                           pjz, pjr, pjh, KT1);
    // 5) layer-1 recurrence -> final output [T,B,H] f32
    init_rec_kernel<<<64, 256, 0, stream>>>(hbuf, hp, bar);
    rec_kernel<<<REC_NWG, 64, REC_LDS_BYTES, stream>>>(
        pjz, pjr, pjh, Urec1, hbuf, hp, rhp, nullptr, (float*)d_out, bar);
}
